// MoE_81492709474999
// MI455X (gfx1250) — matrix operand order repaired and checked by reading, compiled, not run on hardware
//
#include <hip/hip_runtime.h>
#include <hip/hip_bf16.h>
#include <cstdint>
#include <cstddef>

// ---------------- vector types for WMMA ----------------
typedef __attribute__((ext_vector_type(16))) __bf16 v16bf;
typedef __attribute__((ext_vector_type(8)))  __bf16 v8bf;
typedef __attribute__((ext_vector_type(8)))  float  v8f;

// ---------------- TDM descriptor types (amdgpu-toolchain clang-23: 6-arg builtin) ----------------
typedef __attribute__((ext_vector_type(4))) unsigned tdm_g0_t;
typedef __attribute__((ext_vector_type(8))) int      tdm_g1_t;
typedef __attribute__((ext_vector_type(4))) int      tdm_g2_t;

#if defined(__HIP_DEVICE_COMPILE__) && __has_builtin(__builtin_amdgcn_tensor_load_to_lds) && \
    __has_builtin(__builtin_amdgcn_s_wait_tensorcnt)
#define MOE_HAS_TDM 1
#else
#define MOE_HAS_TDM 0
#endif

// ---------------- problem constants ----------------
#define MOE_B   16
#define MOE_T   8192
#define MOE_C   256
#define MOE_E   8
#define MOE_CAP 1280
#define MOE_DFF 1024

#define BM        64                 // rows (tokens) per block
#define XS_STRIDE (MOE_C + 8)        // 264, LDS stride for x tile (bf16)
#define H_STRIDE  (MOE_DFF + 8)      // 1032, LDS stride for hidden tiles (bf16)
#define WC_KPAD   (MOE_C + 8)        // transposed-weight K stride when K = C
#define WD_KPAD   (MOE_DFF + 8)      // transposed-weight K stride when K = DFF

// ---------------- small helpers ----------------
__device__ __forceinline__ v8f zero_v8f() {
  v8f z;
#pragma unroll
  for (int i = 0; i < 8; ++i) z[i] = 0.0f;
  return z;
}

__device__ __forceinline__ v16bf cat16(v8bf lo, v8bf hi) {
  return __builtin_shufflevector(lo, hi, 0,1,2,3,4,5,6,7,8,9,10,11,12,13,14,15);
}

// A-fragment (16x32 bf16, MxK) per ISA 7.12.2:
// lanes 0-15 row M=r hold K {k0..k0+7, k0+16..k0+23}; lanes 16-31 hold the +8 halves.
__device__ __forceinline__ v16bf load_frag_a(const __bf16* base, int stride,
                                             int m0, int r, int half, int k0) {
  const __bf16* p = base + (size_t)(m0 + r) * stride + k0 + half * 8;
  return cat16(*(const v8bf*)p, *(const v8bf*)(p + 16));
}

// B-fragment (32x16 bf16, KxN) from [N][Kpad]-transposed weights:
// lane r = column n; lanes 0-15 hold K {k0..k0+7, k0+16..k0+23}; lanes 16-31 hold the +8 halves.
__device__ __forceinline__ v16bf load_frag_b(const __bf16* wt, int kpad,
                                             int n, int half, int k0) {
  const __bf16* p = wt + (size_t)n * kpad + k0 + half * 8;
  return cat16(*(const v8bf*)p, *(const v8bf*)(p + 16));
}

__device__ __forceinline__ v8f wmma_bf16(v16bf a, v16bf b, v8f c) {
  return __builtin_amdgcn_wmma_f32_16x16x32_bf16(false, a, false, b, (short)0, c,
                                                 false, false);
}

#if MOE_HAS_TDM
// Build a D# (ISA cdna5 §8.3-8.4) and issue one TENSOR_LOAD_TO_LDS.
// flags0: extra group0 flag bits (bit31 = gather_mode, bit30 = 32-bit indices).
// data_size is fixed to 4 bytes (code 2). 2D tile: tile_d0 x tile_d1, row stride stride0.
__device__ __forceinline__ void tdm_issue(unsigned flags0, unsigned lds_off,
                                          unsigned long long ga,
                                          unsigned tensor_d0, unsigned tensor_d1,
                                          unsigned tile_d0, unsigned tile_d1,
                                          unsigned stride0, tdm_g2_t g2, tdm_g2_t g3) {
  tdm_g0_t g0;
  g0[0] = 1u | flags0;                                        // count=1 (+gather bits)
  g0[1] = lds_off;                                            // lds_addr[31:0]
  g0[2] = (unsigned)ga;                                       // global_addr[31:0]
  g0[3] = ((unsigned)(ga >> 32) & 0x01FFFFFFu) | 0x80000000u; // addr[56:32] | type=2
  tdm_g1_t g1;
  g1[0] = (int)(2u << 16);                                    // wg_mask=0, data_size=4B
  g1[1] = (int)((tensor_d0 & 0xFFFFu) << 16);                 // tensor_dim0[15:0]
  g1[2] = (int)(((tensor_d0 >> 16) & 0xFFFFu) | ((tensor_d1 & 0xFFFFu) << 16));
  g1[3] = (int)(((tensor_d1 >> 16) & 0xFFFFu) | ((tile_d0 & 0xFFFFu) << 16));
  g1[4] = (int)(tile_d1 & 0xFFFFu);                           // tile_dim1, tile_dim2=0
  g1[5] = (int)stride0;                                       // tensor_dim0_stride[31:0]
  g1[6] = 0;
  g1[7] = 0;
  tdm_g1_t g4;                                                // 5th operand (clang-23 form)
#pragma unroll
  for (int i = 0; i < 8; ++i) g4[i] = 0;
  __builtin_amdgcn_tensor_load_to_lds(g0, g1, g2, g3, g4, 0);
}
#endif

// 64 x 1024 GEMM stage: As[64,KDIM](LDS bf16) @ WT[N=1024][KPADW](global bf16),
// +bias, ReLU, write bf16 into Ds(LDS). 8 waves, wave tile 64x64, 2 N-passes.
template<int KDIM, int KPADW>
__device__ __forceinline__ void gemm64_stage(const __bf16* __restrict__ As, int asStride,
                                             const __bf16* __restrict__ WT,
                                             const float* __restrict__ bias,
                                             __bf16* __restrict__ Ds, int dsStride,
                                             int wave, int half, int r) {
#pragma unroll 1
  for (int pass = 0; pass < 2; ++pass) {
    const int n0w = pass * 512 + wave * 64;
    v8f acc[4][4];
#pragma unroll
    for (int mi = 0; mi < 4; ++mi)
#pragma unroll
      for (int ni = 0; ni < 4; ++ni) acc[mi][ni] = zero_v8f();

#pragma unroll 1
    for (int k0 = 0; k0 < KDIM; k0 += 32) {
      v16bf a[4], b[4];
#pragma unroll
      for (int mi = 0; mi < 4; ++mi)
        a[mi] = load_frag_a(As, asStride, mi * 16, r, half, k0);
#pragma unroll
      for (int ni = 0; ni < 4; ++ni) {
        const int n = n0w + ni * 16 + r;
        b[ni] = load_frag_b(WT, KPADW, n, half, k0);
        if (k0 + 32 < KDIM)  // global_prefetch_b8 of next K-slice
          __builtin_prefetch(WT + (size_t)n * KPADW + (k0 + 32), 0, 1);
      }
#pragma unroll
      for (int mi = 0; mi < 4; ++mi)
#pragma unroll
        for (int ni = 0; ni < 4; ++ni)
          acc[mi][ni] = wmma_bf16(a[mi], b[ni], acc[mi][ni]);
    }

    // epilogue: D layout (ISA 7.12.2): VGPR i, lanes 0-15 -> M=i, lanes 16-31 -> M=8+i; N=lane&15
#pragma unroll
    for (int ni = 0; ni < 4; ++ni) {
      const int n = n0w + ni * 16 + r;
      const float bv = bias[n];
#pragma unroll
      for (int mi = 0; mi < 4; ++mi) {
#pragma unroll
        for (int i = 0; i < 8; ++i) {
          const int m = mi * 16 + half * 8 + i;
          float v = acc[mi][ni][i] + bv;
          v = v > 0.0f ? v : 0.0f;
          Ds[m * dsStride + n] = (__bf16)v;
        }
      }
    }
  }
}

// ================= fused router: relu(x@W1+b1), relu(@W2+b2), @W3+b3, softmax, top2 ==========
__global__ __launch_bounds__(256) void moe_router_kernel(
    const float* __restrict__ x,
    const __bf16* __restrict__ W1T, const float* __restrict__ b1,
    const __bf16* __restrict__ W2T, const float* __restrict__ b2,
    const float* __restrict__ W3,  const float* __restrict__ b3,
    int* __restrict__ route_id, float* __restrict__ route_gate) {
  extern __shared__ char smem[];
  __bf16* xs = (__bf16*)smem;                       // 64 x 264
  __bf16* h1 = xs + BM * XS_STRIDE;                 // 64 x 1032
  __bf16* h2 = h1 + BM * H_STRIDE;                  // 64 x 1032
  float*  lg = (float*)(h2 + BM * H_STRIDE);        // 64 x 8

  const int tid  = threadIdx.x;
  const int wave = tid >> 5;
  const int lane = tid & 31;
  const int half = lane >> 4;
  const int r    = lane & 15;
  const int row0 = blockIdx.x * BM;                 // global token row

  const float* xrow = x + (size_t)row0 * MOE_C;

#if MOE_HAS_TDM
  // Stage the contiguous 64x256 f32 tile via the Tensor Data Mover into the
  // (currently unused) h2 region, then convert to bf16 in xs.
  float* stage = (float*)h2;                        // 64 KB < 129 KB region
  if (wave == 0) {
    tdm_g2_t zz;
    zz[0] = 0; zz[1] = 0; zz[2] = 0; zz[3] = 0;
    tdm_issue(/*flags0=*/0u,
              (unsigned)(size_t)(void*)stage,
              (unsigned long long)(size_t)xrow,
              /*tensor_d0=*/MOE_C, /*tensor_d1=*/BM,
              /*tile_d0=*/MOE_C,   /*tile_d1=*/BM,
              /*stride0=*/MOE_C, zz, zz);
    __builtin_amdgcn_s_wait_tensorcnt(0);
  }
  __syncthreads();
  for (int v = tid; v < BM * MOE_C / 4; v += 256) {
    const int rr = (v * 4) / MOE_C, cc = (v * 4) % MOE_C;
    const float4 f = *(const float4*)(stage + rr * MOE_C + cc);
    __bf16* p = xs + rr * XS_STRIDE + cc;
    p[0] = (__bf16)f.x; p[1] = (__bf16)f.y; p[2] = (__bf16)f.z; p[3] = (__bf16)f.w;
  }
#else
  for (int v = tid; v < BM * MOE_C / 4; v += 256) {
    const int rr = (v * 4) / MOE_C, cc = (v * 4) % MOE_C;
    const float4 f = *(const float4*)(xrow + (size_t)rr * MOE_C + cc);
    __bf16* p = xs + rr * XS_STRIDE + cc;
    p[0] = (__bf16)f.x; p[1] = (__bf16)f.y; p[2] = (__bf16)f.z; p[3] = (__bf16)f.w;
  }
#endif
  __syncthreads();

  gemm64_stage<MOE_C,  WC_KPAD>(xs, XS_STRIDE, W1T, b1, h1, H_STRIDE, wave, half, r);
  __syncthreads();
  gemm64_stage<MOE_DFF, WD_KPAD>(h1, H_STRIDE,  W2T, b2, h2, H_STRIDE, wave, half, r);
  __syncthreads();

  // GEMM3 (N=8): each thread does 2 (token, expert) dot products over DFF
  for (int p = tid; p < BM * MOE_E; p += 256) {
    const int t = p >> 3, e = p & 7;
    float s = b3[e];
    const __bf16* hp = h2 + t * H_STRIDE;
    for (int k = 0; k < MOE_DFF; k += 8) {
      const v8bf hv = *(const v8bf*)(hp + k);
#pragma unroll
      for (int j = 0; j < 8; ++j) s += (float)hv[j] * W3[(k + j) * MOE_E + e];
    }
    lg[t * MOE_E + e] = s;
  }
  __syncthreads();

  // softmax + top-2 per token
  if (tid < BM) {
    const float* L = lg + tid * MOE_E;
    float mx = L[0];
#pragma unroll
    for (int j = 1; j < MOE_E; ++j) mx = L[j] > mx ? L[j] : mx;
    float pr[MOE_E], s = 0.0f;
#pragma unroll
    for (int j = 0; j < MOE_E; ++j) { pr[j] = __expf(L[j] - mx); s += pr[j]; }
    const float inv = 1.0f / s;
    int i0 = 0; float p0 = pr[0];
#pragma unroll
    for (int j = 1; j < MOE_E; ++j) if (pr[j] > p0) { p0 = pr[j]; i0 = j; }
    int i1 = -1; float p1 = -1.0f;
#pragma unroll
    for (int j = 0; j < MOE_E; ++j) if (j != i0 && pr[j] > p1) { p1 = pr[j]; i1 = j; }
    const size_t row = (size_t)row0 + tid;
    route_id[row * 2 + 0]   = i0;
    route_id[row * 2 + 1]   = i1;
    route_gate[row * 2 + 0] = p0 * inv;
    route_gate[row * 2 + 1] = p1 * inv;
  }
}

// ================= routing scan: rank, capacity subsample, gather lists ==========
__global__ __launch_bounds__(256) void moe_route_scan_kernel(
    const int* __restrict__ route_id, const float* __restrict__ route_gate,
    int* __restrict__ gather_idx, float* __restrict__ gather_gate,
    int* __restrict__ cap_cnt) {
  const int be = blockIdx.x;            // b*E + e
  const int e  = be % MOE_E;
  const int b  = be / MOE_E;
  const int tid = threadIdx.x, wave = tid >> 5, lane = tid & 31;
  const size_t base = (size_t)b * MOE_T;

  __shared__ int red[256];
  __shared__ int wsum[8];

  // pass 1: total count for this (row, expert)
  int local = 0;
  for (int t = tid; t < MOE_T; t += 256) {
    const size_t rw = (base + t) * 2;
    local += (route_id[rw] == e) || (route_id[rw + 1] == e);
  }
  red[tid] = local;
  __syncthreads();
  for (int s = 128; s > 0; s >>= 1) {
    if (tid < s) red[tid] += red[tid + s];
    __syncthreads();
  }
  const int cnt = red[0];
  if (tid == 0) cap_cnt[be] = cnt < MOE_CAP ? cnt : MOE_CAP;
  __syncthreads();

  // pass 2: wave32 ballot scan -> rank; linspace-subsample keep/slot
  const long denCap = MOE_CAP - 1;
  const long denCnt = (long)cnt - 1;
  int running = 0;
  for (int t0 = 0; t0 < MOE_T; t0 += 256) {
    const int t = t0 + tid;
    const size_t rw = (base + t) * 2;
    const int i0 = route_id[rw], i1 = route_id[rw + 1];
    const bool sel = (i0 == e) || (i1 == e);
    const float w = (i0 == e) ? route_gate[rw] : ((i1 == e) ? route_gate[rw + 1] : 0.0f);

    const unsigned long long mask = __ballot(sel);
    const int lrank = __popcll(mask & ((1ull << lane) - 1ull));
    const int wtot  = __popcll(mask);
    if (lane == 0) wsum[wave] = wtot;
    __syncthreads();
    int off = 0, tot = 0;
#pragma unroll
    for (int wv = 0; wv < 8; ++wv) {
      const int s = wsum[wv];
      tot += s;
      if (wv < wave) off += s;
    }
    if (sel) {
      const int rank = running + off + lrank;
      bool kept; int slot;
      if (cnt <= MOE_CAP) {
        kept = true; slot = rank;
      } else {
        // smallest j with floor(j*(cnt-1)/(CAP-1)) == rank
        const long j = ((long)rank * denCap + denCnt - 1) / denCnt;
        kept = (j * denCnt < (long)(rank + 1) * denCap) && (j < MOE_CAP);
        slot = (int)j;
      }
      if (kept) {
        gather_idx[(size_t)be * MOE_CAP + slot]  = t;
        gather_gate[(size_t)be * MOE_CAP + slot] = w;
      }
    }
    running += tot;
    __syncthreads();
  }
}

// ================= fused expert FFN: gather -> relu(@Wi+bi) -> @Wo+bo -> gated scatter-add ======
__global__ __launch_bounds__(256) void moe_expert_kernel(
    const float* __restrict__ x,
    const __bf16* __restrict__ WiT, const float* __restrict__ ebi,
    const __bf16* __restrict__ WoT, const float* __restrict__ ebo,
    const int* __restrict__ gather_idx, const float* __restrict__ gather_gate,
    const int* __restrict__ cap_cnt, float* __restrict__ out) {
  const int nsb = MOE_CAP / BM;        // 20 slot-blocks per (b,e)
  const int blk = blockIdx.x;
  const int sb  = blk % nsb;
  const int be  = blk / nsb;
  const int e   = be % MOE_E;
  const int b   = be / MOE_E;
  const int cnt = cap_cnt[be];
  const int s0  = sb * BM;
  if (s0 >= cnt) return;               // uniform per block

  extern __shared__ char smem[];
  __bf16* xs = (__bf16*)smem;                  // 64 x 264
  __bf16* h  = xs + BM * XS_STRIDE;            // 64 x 1032
  int*    tl = (int*)(h + BM * H_STRIDE);      // 64 token ids
  float*  gl = (float*)(tl + BM);              // 64 gates

  const int tid  = threadIdx.x;
  const int wave = tid >> 5;
  const int lane = tid & 31;
  const int half = lane >> 4;
  const int r    = lane & 15;

  if (tid < BM) {
    const int slot = s0 + tid;
    if (slot < cnt) {
      tl[tid] = gather_idx[(size_t)be * MOE_CAP + slot];
      gl[tid] = gather_gate[(size_t)be * MOE_CAP + slot];
    } else {
      tl[tid] = 0;
      gl[tid] = 0.0f;
    }
  }
  __syncthreads();

#if MOE_HAS_TDM
  // TDM gather mode: 4 descriptors x 16 indexed rows of 256 f32 each,
  // staged into the (currently unused) h region, then converted to bf16.
  float* stage = (float*)h;                    // 64 KB < 129 KB region
  if (wave == 0) {
    const unsigned long long gbase =
        (unsigned long long)(size_t)(x + (size_t)b * MOE_T * MOE_C);
#pragma unroll
    for (int d = 0; d < 4; ++d) {
      const int* tp = tl + d * 16;
      tdm_g2_t g2, g3;
#pragma unroll
      for (int wv = 0; wv < 4; ++wv) {
        g2[wv] = (tp[2 * wv] & 0xFFFF)     | ((tp[2 * wv + 1] & 0xFFFF) << 16);
        g3[wv] = (tp[8 + 2 * wv] & 0xFFFF) | ((tp[9 + 2 * wv] & 0xFFFF) << 16);
      }
      tdm_issue(/*flags0=*/0x80000000u /* gather_mode, 16-bit indices */,
                (unsigned)(size_t)(void*)(stage + d * 16 * MOE_C),
                gbase,
                /*tensor_d0=*/MOE_C, /*tensor_d1=*/MOE_T,
                /*tile_d0=*/MOE_C,   /*tile_d1=*/16 /* #indices */,
                /*stride0=*/MOE_C, g2, g3);
    }
    __builtin_amdgcn_s_wait_tensorcnt(0);
  }
  __syncthreads();
  for (int v = tid; v < BM * MOE_C / 4; v += 256) {
    const int rr = (v * 4) / MOE_C, cc = (v * 4) % MOE_C;
    const float4 f = *(const float4*)(stage + rr * MOE_C + cc);
    __bf16* p = xs + rr * XS_STRIDE + cc;
    p[0] = (__bf16)f.x; p[1] = (__bf16)f.y; p[2] = (__bf16)f.z; p[3] = (__bf16)f.w;
  }
#else
  for (int v = tid; v < BM * MOE_C / 4; v += 256) {
    const int rr = (v * 4) / MOE_C, cc = (v * 4) % MOE_C;
    const float4 f = *(const float4*)(x + ((size_t)b * MOE_T + tl[rr]) * MOE_C + cc);
    __bf16* p = xs + rr * XS_STRIDE + cc;
    p[0] = (__bf16)f.x; p[1] = (__bf16)f.y; p[2] = (__bf16)f.z; p[3] = (__bf16)f.w;
  }
#endif
  __syncthreads();

  // GEMM1: [64,256] @ WiT[e] -> relu -> h (bf16 LDS)
  gemm64_stage<MOE_C, WC_KPAD>(xs, XS_STRIDE,
                               WiT + (size_t)e * MOE_DFF * WC_KPAD,
                               ebi + (size_t)e * MOE_DFF,
                               h, H_STRIDE, wave, half, r);
  __syncthreads();

  // GEMM2: [64,1024] @ WoT[e] (N=256) + gated atomic scatter
  {
    const __bf16* WT = WoT + (size_t)e * MOE_C * WD_KPAD;
    const int n0w = wave * 32;
    v8f acc[4][2];
#pragma unroll
    for (int mi = 0; mi < 4; ++mi)
#pragma unroll
      for (int ni = 0; ni < 2; ++ni) acc[mi][ni] = zero_v8f();

#pragma unroll 1
    for (int k0 = 0; k0 < MOE_DFF; k0 += 32) {
      v16bf a[4], bfr[2];
#pragma unroll
      for (int mi = 0; mi < 4; ++mi)
        a[mi] = load_frag_a(h, H_STRIDE, mi * 16, r, half, k0);
#pragma unroll
      for (int ni = 0; ni < 2; ++ni) {
        const int n = n0w + ni * 16 + r;
        bfr[ni] = load_frag_b(WT, WD_KPAD, n, half, k0);
        if (k0 + 32 < MOE_DFF)
          __builtin_prefetch(WT + (size_t)n * WD_KPAD + (k0 + 32), 0, 1);
      }
#pragma unroll
      for (int mi = 0; mi < 4; ++mi)
#pragma unroll
        for (int ni = 0; ni < 2; ++ni)
          acc[mi][ni] = wmma_bf16(a[mi], bfr[ni], acc[mi][ni]);
    }

    const float* ebop = ebo + (size_t)e * MOE_C;
#pragma unroll
    for (int ni = 0; ni < 2; ++ni) {
      const int n = n0w + ni * 16 + r;
      const float bv = ebop[n];
#pragma unroll
      for (int mi = 0; mi < 4; ++mi) {
#pragma unroll
        for (int i = 0; i < 8; ++i) {
          const int m = mi * 16 + half * 8 + i;
          const float g = gl[m];
          if (g != 0.0f) {
            const float v = g * (acc[mi][ni][i] + bv);
            atomicAdd(out + ((size_t)b * MOE_T + tl[m]) * MOE_C + n, v);
          }
        }
      }
    }
  }
}

// ================= weight transpose + fp32 -> bf16 convert ==========
__global__ __launch_bounds__(256) void moe_wtrans_kernel(
    const float* __restrict__ src, __bf16* __restrict__ dst,
    int K, int N, int Kpad) {
  const size_t idx = (size_t)blockIdx.x * 256 + threadIdx.x;
  if (idx >= (size_t)K * N) return;
  const int k = (int)(idx / N);
  const int n = (int)(idx % N);
  dst[(size_t)n * Kpad + k] = (__bf16)src[idx];
}

// ================= host launch ==========
extern "C" void kernel_launch(void* const* d_in, const int* in_sizes, int n_in,
                              void* d_out, int out_size, void* d_ws, size_t ws_size,
                              hipStream_t stream) {
  const float* x   = (const float*)d_in[0];
  const float* rW1 = (const float*)d_in[1];
  const float* rb1 = (const float*)d_in[2];
  const float* rW2 = (const float*)d_in[3];
  const float* rb2 = (const float*)d_in[4];
  const float* rW3 = (const float*)d_in[5];
  const float* rb3 = (const float*)d_in[6];
  const float* eWi = (const float*)d_in[7];
  const float* ebi = (const float*)d_in[8];
  const float* eWo = (const float*)d_in[9];
  const float* ebo = (const float*)d_in[10];
  float* out = (float*)d_out;
  (void)n_in; (void)in_sizes; (void)ws_size;

  // workspace partition
  char* w = (char*)d_ws;
  auto alloc = [&](size_t bytes) -> char* {
    char* p = w;
    w += (bytes + 255) & ~(size_t)255;
    return p;
  };
  int*    route_id    = (int*)   alloc((size_t)MOE_B * MOE_T * 2 * sizeof(int));
  float*  route_gate  = (float*) alloc((size_t)MOE_B * MOE_T * 2 * sizeof(float));
  int*    gather_idx  = (int*)   alloc((size_t)MOE_B * MOE_E * MOE_CAP * sizeof(int));
  float*  gather_gate = (float*) alloc((size_t)MOE_B * MOE_E * MOE_CAP * sizeof(float));
  int*    cap_cnt     = (int*)   alloc((size_t)MOE_B * MOE_E * sizeof(int));
  __bf16* W1T         = (__bf16*)alloc((size_t)MOE_DFF * WC_KPAD * sizeof(__bf16));
  __bf16* W2T         = (__bf16*)alloc((size_t)MOE_DFF * WD_KPAD * sizeof(__bf16));
  __bf16* WiT         = (__bf16*)alloc((size_t)MOE_E * MOE_DFF * WC_KPAD * sizeof(__bf16));
  __bf16* WoT         = (__bf16*)alloc((size_t)MOE_E * MOE_C * WD_KPAD * sizeof(__bf16));

  (void)hipMemsetAsync(d_out, 0, (size_t)out_size * sizeof(float), stream);

  // transpose + bf16-convert weights (L2-resident afterwards)
  {
    const int nW1 = (MOE_C * MOE_DFF + 255) / 256;
    moe_wtrans_kernel<<<nW1, 256, 0, stream>>>(rW1, W1T, MOE_C, MOE_DFF, WC_KPAD);
    const int nW2 = (MOE_DFF * MOE_DFF + 255) / 256;
    moe_wtrans_kernel<<<nW2, 256, 0, stream>>>(rW2, W2T, MOE_DFF, MOE_DFF, WD_KPAD);
    for (int e = 0; e < MOE_E; ++e) {
      moe_wtrans_kernel<<<nW1, 256, 0, stream>>>(
          eWi + (size_t)e * MOE_C * MOE_DFF, WiT + (size_t)e * MOE_DFF * WC_KPAD,
          MOE_C, MOE_DFF, WC_KPAD);
      moe_wtrans_kernel<<<nW1, 256, 0, stream>>>(
          eWo + (size_t)e * MOE_DFF * MOE_C, WoT + (size_t)e * MOE_C * WD_KPAD,
          MOE_DFF, MOE_C, WD_KPAD);
    }
  }

  // fused router MLP + softmax + top-2 (all hidden activations live in LDS)
  {
    const size_t shR = (size_t)(BM * XS_STRIDE + 2 * BM * H_STRIDE) * sizeof(__bf16)
                     + (size_t)BM * MOE_E * sizeof(float);
    moe_router_kernel<<<(MOE_B * MOE_T) / BM, 256, shR, stream>>>(
        x, W1T, rb1, W2T, rb2, rW3, rb3, route_id, route_gate);
  }

  // ranks + capacity subsample -> gather lists
  moe_route_scan_kernel<<<MOE_B * MOE_E, 256, 0, stream>>>(
      route_id, route_gate, gather_idx, gather_gate, cap_cnt);

  // fused expert FFN with gather + gated atomic scatter
  {
    const size_t shE = (size_t)(BM * XS_STRIDE + BM * H_STRIDE) * sizeof(__bf16)
                     + (size_t)BM * (sizeof(int) + sizeof(float));
    moe_expert_kernel<<<MOE_B * MOE_E * (MOE_CAP / BM), 256, shE, stream>>>(
        x, WiT, ebi, WoT, ebo, gather_idx, gather_gate, cap_cnt, out);
  }
}
